// MPNNModel_23373212024952
// MI455X (gfx1250) — matrix operand order repaired and checked by reading, not compiled, not run on hardware
//
#include <hip/hip_runtime.h>

#define N_NODES_C 50000
#define N_EDGES_C 800000
#define IN_DIM_C  128
#define EMB_C     64
#define EDGE_DIM_C 6
#define N_LAYERS_C 4
#define K1PAD     160            // 2*EMB + EDGE_DIM = 134 padded to 5*32
#define WAVES     4
#define TPB       (WAVES * 32)

typedef _Float16 h16_t;
typedef __attribute__((ext_vector_type(16))) _Float16 v16h;
typedef __attribute__((ext_vector_type(8)))  _Float16 v8h;
typedef __attribute__((ext_vector_type(8)))  float    v8f;
typedef int v4i_vs __attribute__((__vector_size__(16)));   // matches builtin prototype

#define DEV static __device__ __forceinline__

// ---- async direct-to-LDS gather (gfx1250 GLOBAL_LOAD_ASYNC_TO_LDS_*) ----
#if defined(__has_builtin)
#  if __has_builtin(__builtin_amdgcn_global_load_async_to_lds_b128)
#    define HAVE_ASYNC_LDS 1
#  endif
#endif

#if defined(HAVE_ASYNC_LDS)
DEV void async_ld_b128(const void* g, void* l) {
    __builtin_amdgcn_global_load_async_to_lds_b128(
        (__attribute__((address_space(1))) v4i_vs*)(uintptr_t)g,
        (__attribute__((address_space(3))) v4i_vs*)(uint32_t)(uintptr_t)l,
        0, 0);
}
#endif

DEV void wait_async0() {
#if defined(__has_builtin)
#  if __has_builtin(__builtin_amdgcn_s_wait_asynccnt)
    __builtin_amdgcn_s_wait_asynccnt(0);
    asm volatile("" ::: "memory");
    return;
#  endif
#endif
    asm volatile("s_wait_asynccnt 0x0" ::: "memory");
}

DEV v8f vzero8() {
    v8f z;
#pragma unroll
    for (int i = 0; i < 8; ++i) z[i] = 0.0f;
    return z;
}

// A-fragment (16x32 f16, ISA 7.12.2): lanes 0-15 row M=lane, K {c+0..7, c+16..23};
// lanes 16-31 row M=lane-16, K {c+8..15, c+24..31}.  row_k0 = &A[lane&15][kchunk*32]
DEV v16h frag_a(const h16_t* row_k0, int lane) {
    const h16_t* p = row_k0 + ((lane & 16) ? 8 : 0);
    v8h lo = *(const v8h*)p;
    v8h hi = *(const v8h*)(p + 16);
    return __builtin_shufflevector(lo, hi, 0,1,2,3,4,5,6,7,8,9,10,11,12,13,14,15);
}

// B-fragment (32x16 f16): lanes 0-15 col N=lane, K {c+0..7, c+16..23}; lanes 16-31 col N=lane-16,
// K {c+8..15, c+24..31}. Weights stored transposed in LDS: wt[n][k].  col_k0 = &wt[N][kchunk*32]
DEV v16h frag_b(const h16_t* col_k0, int lane) {
    const h16_t* p = col_k0 + ((lane & 16) ? 8 : 0);
    v8h lo = *(const v8h*)p;
    v8h hi = *(const v8h*)(p + 16);
    return __builtin_shufflevector(lo, hi, 0,1,2,3,4,5,6,7,8,9,10,11,12,13,14,15);
}

DEV v8f wmma(v16h a, v16h b, v8f c) {
    return __builtin_amdgcn_wmma_f32_16x16x32_f16(false, a, false, b, (short)0, c, false, false);
}

// ------------------------------------------------------------------
// h = x @ W_in + b ;  also writes f16 copy of h for fast edge gathers
// ------------------------------------------------------------------
__global__ __launch_bounds__(TPB) void k_lin_in(const float* __restrict__ x,
                                                const float* __restrict__ w,
                                                const float* __restrict__ b,
                                                float* __restrict__ h32,
                                                h16_t* __restrict__ h16,
                                                int n_nodes) {
    __shared__ __align__(16) h16_t wt[EMB_C][IN_DIM_C];     // transposed [n][k]
    __shared__ float bs[EMB_C];
    __shared__ __align__(16) h16_t a_lds[WAVES][16][IN_DIM_C];

    int t = threadIdx.x;
    for (int i = t; i < EMB_C * IN_DIM_C; i += TPB) {
        int n = i / IN_DIM_C, k = i % IN_DIM_C;
        wt[n][k] = (h16_t)w[k * EMB_C + n];
    }
    if (t < EMB_C) bs[t] = b[t];
    __syncthreads();

    int wave = t >> 5, lane = t & 31;
    int row = lane & 15;
    int mbase = (lane & 16) ? 8 : 0;
    h16_t (*A)[IN_DIM_C] = a_lds[wave];

    int ntiles = n_nodes / 16;
    for (int tile = blockIdx.x * WAVES + wave; tile < ntiles; tile += gridDim.x * WAVES) {
        int m0 = tile * 16;
        for (int e = 0; e < 16; ++e) {
            float4 v = ((const float4*)(x + (size_t)(m0 + e) * IN_DIM_C))[lane];
            A[e][lane * 4 + 0] = (h16_t)v.x;
            A[e][lane * 4 + 1] = (h16_t)v.y;
            A[e][lane * 4 + 2] = (h16_t)v.z;
            A[e][lane * 4 + 3] = (h16_t)v.w;
        }

        v8f acc[4];
#pragma unroll
        for (int nt = 0; nt < 4; ++nt) acc[nt] = vzero8();
#pragma unroll
        for (int kc = 0; kc < IN_DIM_C / 32; ++kc) {
            v16h a = frag_a(&A[row][kc * 32], lane);
#pragma unroll
            for (int nt = 0; nt < 4; ++nt) {
                v16h bb = frag_b(&wt[nt * 16 + row][kc * 32], lane);
                acc[nt] = wmma(a, bb, acc[nt]);
            }
        }

#pragma unroll
        for (int nt = 0; nt < 4; ++nt) {
            int n = nt * 16 + row;
            float bias = bs[n];
#pragma unroll
            for (int r = 0; r < 8; ++r) {
                float v = acc[nt][r] + bias;
                size_t idx = (size_t)(m0 + mbase + r) * EMB_C + n;
                h32[idx] = v;
                h16[idx] = (h16_t)v;
            }
        }
    }
}

// ------------------------------------------------------------------
// fused per-edge: gather [h[dst],h[src],e_attr] -> MLP(134->64->64, relu)
//                 -> atomic scatter-add into agg[dst]
// ------------------------------------------------------------------
__global__ __launch_bounds__(TPB) void k_edge(const h16_t* __restrict__ h16g,
                                              const int* __restrict__ src,
                                              const int* __restrict__ dst,
                                              const float* __restrict__ eattr,
                                              const float* __restrict__ w1,
                                              const float* __restrict__ b1,
                                              const float* __restrict__ w2,
                                              const float* __restrict__ b2,
                                              float* __restrict__ agg,
                                              int n_edges) {
    __shared__ __align__(16) h16_t wt1[EMB_C][K1PAD];       // transposed, K padded w/ zeros
    __shared__ __align__(16) h16_t wt2[EMB_C][EMB_C];
    __shared__ float bs1[EMB_C], bs2[EMB_C];
    __shared__ __align__(16) h16_t a_lds[WAVES][16][K1PAD];
    __shared__ __align__(16) h16_t hid[WAVES][16][EMB_C];

    int t = threadIdx.x;
    for (int i = t; i < EMB_C * K1PAD; i += TPB) {
        int n = i / K1PAD, k = i % K1PAD;
        wt1[n][k] = (k < 2 * EMB_C + EDGE_DIM_C) ? (h16_t)w1[k * EMB_C + n] : (h16_t)0.0f;
    }
    for (int i = t; i < EMB_C * EMB_C; i += TPB) {
        int n = i / EMB_C, k = i % EMB_C;
        wt2[n][k] = (h16_t)w2[k * EMB_C + n];
    }
    if (t < EMB_C) { bs1[t] = b1[t]; bs2[t] = b2[t]; }
    __syncthreads();

    int wave = t >> 5, lane = t & 31;
    int row = lane & 15;
    int mbase = (lane & 16) ? 8 : 0;
    h16_t (*A)[K1PAD] = a_lds[wave];
    h16_t (*H)[EMB_C] = hid[wave];

    // zero K padding cols 134..159 once (never overwritten inside tile loop)
    for (int idx = lane; idx < 16 * (K1PAD - 134); idx += 32) {
        int e = idx / (K1PAD - 134), j = idx % (K1PAD - 134);
        A[e][134 + j] = (h16_t)0.0f;
    }

    int ntiles = n_edges / 16;
    for (int tile = blockIdx.x * WAVES + wave; tile < ntiles; tile += gridDim.x * WAVES) {
        int e0 = tile * 16;

        // ---- gather h[dst] (cols 0..63), h[src] (cols 64..127) ----
#if defined(HAVE_ASYNC_LDS)
        // 8 lanes move one 128B node row via b128; 2 edges x {dst,src} per issue
        for (int e2 = 0; e2 < 16; e2 += 2) {
            int which  = lane >> 3;              // 0..3
            int e      = e2 + (which >> 1);
            int srcsel = which & 1;              // 0: dst row, 1: src row
            int node   = (srcsel ? src : dst)[e0 + e];
            const h16_t* g = h16g + (size_t)node * EMB_C + (lane & 7) * 8;
            h16_t* l = &A[e][srcsel * EMB_C + (lane & 7) * 8];
            async_ld_b128(g, l);
        }
#else
        for (int e = 0; e < 16; ++e) {
            int d = dst[e0 + e];
            int s = src[e0 + e];
            const uint32_t* hd = (const uint32_t*)(h16g + (size_t)d * EMB_C);
            const uint32_t* hs = (const uint32_t*)(h16g + (size_t)s * EMB_C);
            uint32_t* rw = (uint32_t*)A[e];
            rw[lane]      = hd[lane];
            rw[32 + lane] = hs[lane];
        }
#endif
        // edge_attr -> cols 128..133
        for (int idx = lane; idx < 16 * EDGE_DIM_C; idx += 32) {
            int e = idx / EDGE_DIM_C, j = idx % EDGE_DIM_C;
            A[e][2 * EMB_C + j] = (h16_t)eattr[(size_t)(e0 + e) * EDGE_DIM_C + j];
        }
#if defined(HAVE_ASYNC_LDS)
        wait_async0();
#endif

        // GEMM 1: [16 x 160] x [160 x 64]
        v8f acc[4];
#pragma unroll
        for (int nt = 0; nt < 4; ++nt) acc[nt] = vzero8();
#pragma unroll
        for (int kc = 0; kc < K1PAD / 32; ++kc) {
            v16h a = frag_a(&A[row][kc * 32], lane);
#pragma unroll
            for (int nt = 0; nt < 4; ++nt) {
                v16h bb = frag_b(&wt1[nt * 16 + row][kc * 32], lane);
                acc[nt] = wmma(a, bb, acc[nt]);
            }
        }

        // bias + relu -> f16 staging for GEMM 2
#pragma unroll
        for (int nt = 0; nt < 4; ++nt) {
            int n = nt * 16 + row;
            float bias = bs1[n];
#pragma unroll
            for (int r = 0; r < 8; ++r) {
                float v = acc[nt][r] + bias;
                H[mbase + r][n] = (h16_t)(v > 0.0f ? v : 0.0f);
            }
        }

        // GEMM 2: [16 x 64] x [64 x 64]
        v8f acc2[4];
#pragma unroll
        for (int nt = 0; nt < 4; ++nt) acc2[nt] = vzero8();
#pragma unroll
        for (int kc = 0; kc < EMB_C / 32; ++kc) {
            v16h a = frag_a(&H[row][kc * 32], lane);
#pragma unroll
            for (int nt = 0; nt < 4; ++nt) {
                v16h bb = frag_b(&wt2[nt * 16 + row][kc * 32], lane);
                acc2[nt] = wmma(a, bb, acc2[nt]);
            }
        }

        // bias + relu + scatter-add (lane owns column n for rows mbase..mbase+7)
        int dn[8];
#pragma unroll
        for (int r = 0; r < 8; ++r) dn[r] = dst[e0 + mbase + r];
#pragma unroll
        for (int nt = 0; nt < 4; ++nt) {
            int n = nt * 16 + row;
            float bias = bs2[n];
#pragma unroll
            for (int r = 0; r < 8; ++r) {
                float v = acc2[nt][r] + bias;
                v = v > 0.0f ? v : 0.0f;
                atomicAdd(&agg[(size_t)dn[r] * EMB_C + n], v);
            }
        }
    }
}

// ------------------------------------------------------------------
// update: u = relu(relu([h,agg]@W1+b1)@W2+b2);  h += u  (f32 + f16 copies)
// ------------------------------------------------------------------
__global__ __launch_bounds__(TPB) void k_upd(float* __restrict__ h32,
                                             h16_t* __restrict__ h16,
                                             const float* __restrict__ agg,
                                             const float* __restrict__ w1,
                                             const float* __restrict__ b1,
                                             const float* __restrict__ w2,
                                             const float* __restrict__ b2,
                                             int n_nodes) {
    __shared__ __align__(16) h16_t wt1[EMB_C][2 * EMB_C];   // transposed [n][k], K=128
    __shared__ __align__(16) h16_t wt2[EMB_C][EMB_C];
    __shared__ float bs1[EMB_C], bs2[EMB_C];
    __shared__ __align__(16) h16_t a_lds[WAVES][16][2 * EMB_C];
    __shared__ __align__(16) h16_t hid[WAVES][16][EMB_C];

    int t = threadIdx.x;
    for (int i = t; i < EMB_C * 2 * EMB_C; i += TPB) {
        int n = i / (2 * EMB_C), k = i % (2 * EMB_C);
        wt1[n][k] = (h16_t)w1[k * EMB_C + n];
    }
    for (int i = t; i < EMB_C * EMB_C; i += TPB) {
        int n = i / EMB_C, k = i % EMB_C;
        wt2[n][k] = (h16_t)w2[k * EMB_C + n];
    }
    if (t < EMB_C) { bs1[t] = b1[t]; bs2[t] = b2[t]; }
    __syncthreads();

    int wave = t >> 5, lane = t & 31;
    int row = lane & 15;
    int mbase = (lane & 16) ? 8 : 0;
    h16_t (*A)[2 * EMB_C] = a_lds[wave];
    h16_t (*H)[EMB_C] = hid[wave];

    int ntiles = n_nodes / 16;
    for (int tile = blockIdx.x * WAVES + wave; tile < ntiles; tile += gridDim.x * WAVES) {
        int m0 = tile * 16;

#if defined(HAVE_ASYNC_LDS)
        // cols 0..63 (already f16): 8 lanes x b128 per node row, 4 rows per issue
        for (int e4 = 0; e4 < 16; e4 += 4) {
            int e = e4 + (lane >> 3);           // 0..3 sub-rows
            const h16_t* g = h16 + (size_t)(m0 + e) * EMB_C + (lane & 7) * 8;
            h16_t* l = &A[e][(lane & 7) * 8];
            async_ld_b128(g, l);
        }
#else
        for (int e = 0; e < 16; ++e) {
            const uint32_t* hd = (const uint32_t*)(h16 + (size_t)(m0 + e) * EMB_C);
            ((uint32_t*)A[e])[lane] = hd[lane];
        }
#endif
        // cols 64..127: agg (f32 -> f16 conversion must pass through VGPRs)
        for (int e = 0; e < 16; ++e) {
            float2 g = ((const float2*)(agg + (size_t)(m0 + e) * EMB_C))[lane];
            A[e][EMB_C + lane * 2 + 0] = (h16_t)g.x;
            A[e][EMB_C + lane * 2 + 1] = (h16_t)g.y;
        }
#if defined(HAVE_ASYNC_LDS)
        wait_async0();
#endif

        v8f acc[4];
#pragma unroll
        for (int nt = 0; nt < 4; ++nt) acc[nt] = vzero8();
#pragma unroll
        for (int kc = 0; kc < (2 * EMB_C) / 32; ++kc) {
            v16h a = frag_a(&A[row][kc * 32], lane);
#pragma unroll
            for (int nt = 0; nt < 4; ++nt) {
                v16h bb = frag_b(&wt1[nt * 16 + row][kc * 32], lane);
                acc[nt] = wmma(a, bb, acc[nt]);
            }
        }

#pragma unroll
        for (int nt = 0; nt < 4; ++nt) {
            int n = nt * 16 + row;
            float bias = bs1[n];
#pragma unroll
            for (int r = 0; r < 8; ++r) {
                float v = acc[nt][r] + bias;
                H[mbase + r][n] = (h16_t)(v > 0.0f ? v : 0.0f);
            }
        }

        v8f acc2[4];
#pragma unroll
        for (int nt = 0; nt < 4; ++nt) acc2[nt] = vzero8();
#pragma unroll
        for (int kc = 0; kc < EMB_C / 32; ++kc) {
            v16h a = frag_a(&H[row][kc * 32], lane);
#pragma unroll
            for (int nt = 0; nt < 4; ++nt) {
                v16h bb = frag_b(&wt2[nt * 16 + row][kc * 32], lane);
                acc2[nt] = wmma(a, bb, acc2[nt]);
            }
        }

#pragma unroll
        for (int nt = 0; nt < 4; ++nt) {
            int n = nt * 16 + row;
            float bias = bs2[n];
#pragma unroll
            for (int r = 0; r < 8; ++r) {
                float v = acc2[nt][r] + bias;
                v = v > 0.0f ? v : 0.0f;
                size_t idx = (size_t)(m0 + mbase + r) * EMB_C + n;
                float nh = h32[idx] + v;          // residual
                h32[idx] = nh;
                h16[idx] = (h16_t)nh;
            }
        }
    }
}

// ------------------------------------------------------------------
__global__ void k_zero(float* __restrict__ p, int n) {
    int i = blockIdx.x * blockDim.x + threadIdx.x;
    if (i < n) p[i] = 0.0f;
}

__global__ __launch_bounds__(256) void k_colsum(const float* __restrict__ h32,
                                                float* __restrict__ colsum,
                                                int n_nodes) {
    __shared__ float sh[EMB_C];
    if (threadIdx.x < EMB_C) sh[threadIdx.x] = 0.0f;
    __syncthreads();
    int col = threadIdx.x & 63;
    int sub = threadIdx.x >> 6;                // 0..3
    float s = 0.0f;
    for (int m = blockIdx.x * 4 + sub; m < n_nodes; m += gridDim.x * 4)
        s += h32[(size_t)m * EMB_C + col];
    atomicAdd(&sh[col], s);
    __syncthreads();
    if (threadIdx.x < EMB_C) atomicAdd(&colsum[threadIdx.x], sh[threadIdx.x]);
}

__global__ void k_pred(const float* __restrict__ colsum,
                       const float* __restrict__ pw,
                       const float* __restrict__ pb,
                       float* __restrict__ out, int n_nodes) {
    if (threadIdx.x == 0) {
        float s = 0.0f;
        for (int c = 0; c < EMB_C; ++c) s += colsum[c] * pw[c];
        out[0] = s + (float)n_nodes * pb[0];
    }
}

// ------------------------------------------------------------------
extern "C" void kernel_launch(void* const* d_in, const int* in_sizes, int n_in,
                              void* d_out, int out_size, void* d_ws, size_t ws_size,
                              hipStream_t stream) {
    (void)in_sizes; (void)n_in; (void)out_size; (void)ws_size;
    const float* x     = (const float*)d_in[0];
    const int*   ei    = (const int*)d_in[1];
    const float* eattr = (const float*)d_in[2];
    const float* lin_w = (const float*)d_in[3];
    const float* lin_b = (const float*)d_in[4];
    const float* mw1   = (const float*)d_in[5];
    const float* mb1   = (const float*)d_in[6];
    const float* mw2   = (const float*)d_in[7];
    const float* mb2   = (const float*)d_in[8];
    const float* uw1   = (const float*)d_in[9];
    const float* ub1   = (const float*)d_in[10];
    const float* uw2   = (const float*)d_in[11];
    const float* ub2   = (const float*)d_in[12];
    const float* pw    = (const float*)d_in[13];
    const float* pb    = (const float*)d_in[14];
    float* out = (float*)d_out;

    char* ws = (char*)d_ws;
    float* h32    = (float*)(ws);                      // 12.8 MB
    h16_t* h16    = (h16_t*)(ws + 12800000);           // 6.4 MB
    float* agg    = (float*)(ws + 19200000);           // 12.8 MB
    float* colsum = (float*)(ws + 32000000);           // 256 B

    const int* src = ei;                // edge_index[0]
    const int* dst = ei + N_EDGES_C;    // edge_index[1]

    const int NODE_BLOCKS = 512;        // persistent, grid-stride over 3125 tiles
    const int EDGE_BLOCKS = 1024;       // persistent, grid-stride over 50000 tiles
    const int K1 = 2 * EMB_C + EDGE_DIM_C;                           // 134

    k_lin_in<<<NODE_BLOCKS, TPB, 0, stream>>>(x, lin_w, lin_b, h32, h16, N_NODES_C);

    for (int l = 0; l < N_LAYERS_C; ++l) {
        k_zero<<<(N_NODES_C * EMB_C + 255) / 256, 256, 0, stream>>>(agg, N_NODES_C * EMB_C);
        k_edge<<<EDGE_BLOCKS, TPB, 0, stream>>>(h16, src, dst, eattr,
            mw1 + (size_t)l * K1 * EMB_C, mb1 + (size_t)l * EMB_C,
            mw2 + (size_t)l * EMB_C * EMB_C, mb2 + (size_t)l * EMB_C,
            agg, N_EDGES_C);
        k_upd<<<NODE_BLOCKS, TPB, 0, stream>>>(h32, h16, agg,
            uw1 + (size_t)l * 2 * EMB_C * EMB_C, ub1 + (size_t)l * EMB_C,
            uw2 + (size_t)l * EMB_C * EMB_C, ub2 + (size_t)l * EMB_C,
            N_NODES_C);
    }

    k_zero<<<1, 64, 0, stream>>>(colsum, EMB_C);
    k_colsum<<<256, 256, 0, stream>>>(h32, colsum, N_NODES_C);
    k_pred<<<1, 32, 0, stream>>>(colsum, pw, pb, out, N_NODES_C);
}
